// PANPooling_88484916232524
// MI455X (gfx1250) — matrix operand order repaired and checked by reading, not compiled, not run on hardware
//
#include <hip/hip_runtime.h>
#include <hip/hip_bf16.h>

typedef __attribute__((ext_vector_type(16))) _Float16 v16h;
typedef __attribute__((ext_vector_type(8)))  float    v8f;

// ---------------------------------------------------------------- utilities
static __device__ __forceinline__ unsigned int enc_desc(float s) {
    // monotone float->uint (ascending), then invert for descending order
    unsigned int ub = __float_as_uint(s);
    unsigned int sb = (ub & 0x80000000u) ? ~ub : (ub | 0x80000000u);
    return ~sb;
}
static __device__ __forceinline__ float dec_desc(unsigned int hi) {
    unsigned int sb = ~hi;
    unsigned int ub = (sb & 0x80000000u) ? (sb & 0x7FFFFFFFu) : ~sb;
    return __uint_as_float(ub);
}

// ---------------------------------------------------------------- kernels
__global__ void pan_zero_f32(float* __restrict__ a, int n) {
    int i = blockIdx.x * blockDim.x + threadIdx.x;
    if (i < n) a[i] = 0.0f;
}

// pad sort keys past N with +max, init node_idx map to -1
__global__ void pan_init(unsigned long long* __restrict__ keys,
                         int* __restrict__ node_idx, int n, int npad) {
    int i = blockIdx.x * blockDim.x + threadIdx.x;
    if (i < npad && i >= n) keys[i] = ~0ull;
    if (i < n) node_idx[i] = -1;
}

// score2[col[e]] += w[e]
__global__ void pan_scatter_deg(const int* __restrict__ col,
                                const float* __restrict__ w,
                                float* __restrict__ score2, int e) {
    int i = blockIdx.x * blockDim.x + threadIdx.x;
    if (i < e) atomicAdd(&score2[col[i]], w[i]);
}

// score = tanh(b0 * (x . p) + b1 * score2); GEMV via WMMA f16->f32.
// 4 waves / block, each wave computes 16 rows. B = p chunk broadcast to all
// 16 columns, so every D column holds the same dot products.
// CT > 0: compile-time channel count (fully unrolled, 2 accumulators).
template <int CT>
__global__ void __launch_bounds__(128)
pan_score_wmma(const float* __restrict__ x, const float* __restrict__ p,
               const float* __restrict__ beta, const float* __restrict__ score2,
               unsigned long long* __restrict__ keys, int n, int c) {
    const int lane = threadIdx.x & 31;
    const int wave = threadIdx.x >> 5;
    const int rowBase = blockIdx.x * 64 + wave * 16;
    const int h16 = lane >> 4;        // lane half (0: lanes 0-15, 1: 16-31)
    const int m   = lane & 15;

    const int cLim = (CT > 0) ? CT : c;
    int rowA = rowBase + m;
    if (rowA >= n) rowA = n - 1;      // clamp; EXEC stays all-ones for WMMA
    const float* __restrict__ xr = x + (size_t)rowA * cLim;

    v8f acc0 = {}, acc1 = {};
#pragma unroll
    for (int cc = 0; cc < cLim / 32; ++cc) {
        const int kb = cc * 32;
        v16h a, b;
        // A 16x32 f16 layout: lane half 0 -> K 0..7 & 16..23, half 1 -> 8..15 & 24..31
        const float* abase = xr + kb + 8 * h16;
        float4 f0 = *(const float4*)(abase);
        float4 f1 = *(const float4*)(abase + 4);
        float4 f2 = *(const float4*)(abase + 16);
        float4 f3 = *(const float4*)(abase + 20);
        a[0]  = (_Float16)f0.x; a[1]  = (_Float16)f0.y;
        a[2]  = (_Float16)f0.z; a[3]  = (_Float16)f0.w;
        a[4]  = (_Float16)f1.x; a[5]  = (_Float16)f1.y;
        a[6]  = (_Float16)f1.z; a[7]  = (_Float16)f1.w;
        a[8]  = (_Float16)f2.x; a[9]  = (_Float16)f2.y;
        a[10] = (_Float16)f2.z; a[11] = (_Float16)f2.w;
        a[12] = (_Float16)f3.x; a[13] = (_Float16)f3.y;
        a[14] = (_Float16)f3.z; a[15] = (_Float16)f3.w;
        // B 32x16 f16 layout: lane half 0 -> K 0..7 & 16..23, half 1 -> 8..15 & 24..31
        const float* bbase = p + kb + 8 * h16;
        float4 g0 = *(const float4*)(bbase);
        float4 g1 = *(const float4*)(bbase + 4);
        float4 g2 = *(const float4*)(bbase + 16);
        float4 g3 = *(const float4*)(bbase + 20);
        b[0]  = (_Float16)g0.x; b[1]  = (_Float16)g0.y;
        b[2]  = (_Float16)g0.z; b[3]  = (_Float16)g0.w;
        b[4]  = (_Float16)g1.x; b[5]  = (_Float16)g1.y;
        b[6]  = (_Float16)g1.z; b[7]  = (_Float16)g1.w;
        b[8]  = (_Float16)g2.x; b[9]  = (_Float16)g2.y;
        b[10] = (_Float16)g2.z; b[11] = (_Float16)g2.w;
        b[12] = (_Float16)g3.x; b[13] = (_Float16)g3.y;
        b[14] = (_Float16)g3.z; b[15] = (_Float16)g3.w;

        if (cc & 1)
            acc1 = __builtin_amdgcn_wmma_f32_16x16x32_f16(
                false, a, false, b, (short)0, acc1, false, false);
        else
            acc0 = __builtin_amdgcn_wmma_f32_16x16x32_f16(
                false, a, false, b, (short)0, acc0, false, false);
    }
    v8f acc = acc0 + acc1;

    // D layout: VGPR r, lanes 0-15 -> M=r (N=lane), lanes 16-31 -> M=8+r.
    // Column 0 lives in lanes 0 and 16.
    if (m == 0) {
        const float b0 = beta[0], b1 = beta[1];
#pragma unroll
        for (int r = 0; r < 8; ++r) {
            int rowOut = rowBase + 8 * h16 + r;
            if (rowOut < n) {
                float s = tanhf(b0 * acc[r] + b1 * score2[rowOut]);
                keys[rowOut] =
                    ((unsigned long long)enc_desc(s) << 32) | (unsigned int)rowOut;
            }
        }
    }
}

// global bitonic compare-exchange stage (stride j >= 1024)
__global__ void pan_bitonic_global(unsigned long long* __restrict__ keys,
                                   int j, int k) {
    int i = blockIdx.x * blockDim.x + threadIdx.x;
    int l = i ^ j;
    if (l > i) {
        unsigned long long a = keys[i], b = keys[l];
        bool up = ((i & k) == 0);
        if ((a > b) == up) { keys[i] = b; keys[l] = a; }
    }
}

// fused bitonic stages for j <= 512 inside one 1024-element LDS tile.
// Tile load goes through the CDNA5 async DMA path (GLOBAL_LOAD_ASYNC_TO_LDS,
// tracked by ASYNCcnt) instead of a VGPR round-trip.
__global__ void __launch_bounds__(512)
pan_bitonic_fused(unsigned long long* __restrict__ keys, int k, int jstart) {
    __shared__ unsigned long long s[1024];
    const int base = blockIdx.x * 1024;
    const int t = threadIdx.x;
#if defined(__gfx1250__)
    {
        // one b128 per lane: lane t fills s[2t..2t+1] from keys[base+2t..]
        unsigned ldsaddr = (unsigned)(size_t)(&s[2 * t]);   // addr[31:0] = LDS offset
        unsigned voff    = (unsigned)(((size_t)base + 2 * t) * 8u);
        asm volatile("global_load_async_to_lds_b128 %0, %1, %2 offset:0"
                     :: "v"(ldsaddr), "v"(voff), "s"(keys)
                     : "memory");
        asm volatile("s_wait_asynccnt 0x0" ::: "memory");
    }
#else
    s[t]       = keys[base + t];
    s[t + 512] = keys[base + t + 512];
#endif
    __syncthreads();
    for (int j = jstart; j > 0; j >>= 1) {
        int idx = ((t & ~(j - 1)) << 1) | (t & (j - 1));
        unsigned long long a = s[idx], b = s[idx + j];
        bool up = (((base + idx) & k) == 0);
        if ((a > b) == up) { s[idx] = b; s[idx + j] = a; }
        __syncthreads();
    }
    keys[base + t]       = s[t];
    keys[base + t + 512] = s[t + 512];
}

// x_pool[i] = x[perm[i]] * score_perm[i]; emit perm, score_perm, node_idx map
__global__ void pan_pool(const float* __restrict__ x,
                         const unsigned long long* __restrict__ keys,
                         float* __restrict__ x_pool, float* __restrict__ perm_out,
                         float* __restrict__ score_out, int* __restrict__ node_idx,
                         int c) {
    const int i = blockIdx.x;      // pooled row
    const int t = threadIdx.x;     // channel
    unsigned long long key = keys[i];
    unsigned int node = (unsigned int)(key & 0xFFFFFFFFull);
    float score = dec_desc((unsigned int)(key >> 32));
    x_pool[(size_t)i * c + t] = x[(size_t)node * c + t] * score;
    if (t == 0) {
        perm_out[i]  = (float)node;
        score_out[i] = score;
        node_idx[node] = i;
    }
}

// filtered/reindexed adjacency: edge_index = [node_idx[col], node_idx[row]]
__global__ void pan_edges(const int* __restrict__ row, const int* __restrict__ col,
                          const float* __restrict__ w, const int* __restrict__ node_idx,
                          float* __restrict__ ei0, float* __restrict__ ei1,
                          float* __restrict__ eattr, int e) {
    int i = blockIdx.x * blockDim.x + threadIdx.x;
    if (i < e) {
        int nr = node_idx[col[i]];
        int nc = node_idx[row[i]];
        bool keep = (nr >= 0) && (nc >= 0);
        ei0[i]   = (float)(keep ? nr : -1);
        ei1[i]   = (float)(keep ? nc : -1);
        eattr[i] = keep ? w[i] : 0.0f;
    }
}

// ---------------------------------------------------------------- launcher
extern "C" void kernel_launch(void* const* d_in, const int* in_sizes, int n_in,
                              void* d_out, int out_size, void* d_ws, size_t ws_size,
                              hipStream_t stream) {
    const float* x      = (const float*)d_in[0];
    const int*   row    = (const int*)  d_in[1];
    const int*   col    = (const int*)  d_in[2];
    const float* ew     = (const float*)d_in[3];
    const float* p      = (const float*)d_in[4];
    const float* beta   = (const float*)d_in[5];

    const int C = in_sizes[4];            // 256
    const int E = in_sizes[1];            // 3,200,000
    const int N = in_sizes[0] / C;        // 100,000
    const int K = (N + 1) / 2;            // ceil(0.5 * N)
    int NPAD = 1; while (NPAD < N) NPAD <<= 1;   // 131072

    // workspace layout
    char* ws = (char*)d_ws;
    unsigned long long* keys = (unsigned long long*)ws;           // NPAD * 8B
    float* score2   = (float*)(ws + (size_t)NPAD * 8);            // N * 4B
    int*   node_idx = (int*)  (ws + (size_t)NPAD * 8 + (size_t)N * 4); // N * 4B

    // output layout (flat concat in return order)
    float* x_pool    = (float*)d_out;                 // K*C
    float* ei0       = x_pool + (size_t)K * C;        // E
    float* ei1       = ei0 + E;                       // E
    float* eattr     = ei1 + E;                       // E
    float* perm_out  = eattr + E;                     // K
    float* score_out = perm_out + K;                  // K

    const int TB = 256;
    auto cdiv = [](int a, int b) { return (a + b - 1) / b; };

    hipLaunchKernelGGL(pan_zero_f32, dim3(cdiv(N, TB)), dim3(TB), 0, stream, score2, N);
    hipLaunchKernelGGL(pan_init, dim3(cdiv(NPAD, TB)), dim3(TB), 0, stream,
                       keys, node_idx, N, NPAD);
    hipLaunchKernelGGL(pan_scatter_deg, dim3(cdiv(E, TB)), dim3(TB), 0, stream,
                       col, ew, score2, E);
    if (C == 256)
        hipLaunchKernelGGL((pan_score_wmma<256>), dim3(cdiv(N, 64)), dim3(128), 0,
                           stream, x, p, beta, score2, keys, N, C);
    else
        hipLaunchKernelGGL((pan_score_wmma<0>), dim3(cdiv(N, 64)), dim3(128), 0,
                           stream, x, p, beta, score2, keys, N, C);

    // bitonic sort of NPAD packed keys (ascending == score-descending)
    for (int k = 2; k <= NPAD; k <<= 1) {
        int j = k >> 1;
        for (; j >= 1024; j >>= 1)
            hipLaunchKernelGGL(pan_bitonic_global, dim3(NPAD / TB), dim3(TB), 0, stream,
                               keys, j, k);
        int jstart = (k >> 1) < 512 ? (k >> 1) : 512;
        hipLaunchKernelGGL(pan_bitonic_fused, dim3(NPAD / 1024), dim3(512), 0, stream,
                           keys, k, jstart);
    }

    hipLaunchKernelGGL(pan_pool, dim3(K), dim3(C), 0, stream,
                       x, keys, x_pool, perm_out, score_out, node_idx, C);
    hipLaunchKernelGGL(pan_edges, dim3(cdiv(E, TB)), dim3(TB), 0, stream,
                       row, col, ew, node_idx, ei0, ei1, eattr, E);
}
